// RecurrentModel_24893630447855
// MI455X (gfx1250) — compile-verified, matrix operand order checked by reading, not run on hardware
//
#include <hip/hip_runtime.h>
#include <hip/hip_bf16.h>

// ---------------------------------------------------------------------------
// LSTM LM for MI455X (gfx1250): bf16 WMMA GEMMs (f32 accumulate) with
// Tensor-Data-Mover (TDM) async global->LDS staging and double buffering.
//   d_in: [0] tokens int32 [T,B]   [1] emb f32 [V,E]
//         [2] W_x f32 [E,4H]       [3] W_h f32 [H,4H]
//         [4] b f32 [4H]           [5] W_h2o f32 [H,C]   [6] b_o f32 [C]
//   d_out: f32 [T,B,C]
// ---------------------------------------------------------------------------

typedef __bf16 bf16;
typedef __attribute__((ext_vector_type(16))) __bf16 v16bf;
typedef __attribute__((ext_vector_type(8)))  __bf16 v8bf;
typedef __attribute__((ext_vector_type(8)))  float  v8f;
typedef __attribute__((ext_vector_type(4)))  unsigned int v4u;
typedef __attribute__((ext_vector_type(4)))  int v4i;
typedef __attribute__((ext_vector_type(8)))  int v8i;

#define T_STEPS 128
#define BATCH   32
#define EMB_D   1024
#define HID_D   1024
#define G4H     4096     // 4*HID
#define NCLS    32000

// GEMM macro-tile
#define BK 32            // k-slice per stage (one WMMA K)
#define BM 64            // block rows
#define BN 256           // block cols
#define ROW_B   80       // LDS row stride bytes: 64B data + 16B pad (bank-conflict-free, 16B aligned)
#define A_TILE_B (BM * ROW_B)              // 5120 B
#define B_TILE_B (BN * ROW_B)              // 20480 B
#define BUF_B    (A_TILE_B + B_TILE_B)     // 25600 B  (x2 double buffer = 50 KB LDS)

#if defined(__gfx1250__) && __has_builtin(__builtin_amdgcn_tensor_load_to_lds) && \
    __has_builtin(__builtin_amdgcn_s_wait_tensorcnt)
#define HAVE_TDM 1
#endif

// ---------------------------------------------------------------------------
// WMMA fragment layout helpers (16-bit A layout; B uses the same pattern over
// columns since B is stored pre-transposed [N,K] row-major).
// Lanes 0-15: K = {0..7, 16..23}; lanes 16-31: K = {8..15, 24..31}.
// ---------------------------------------------------------------------------
__device__ __forceinline__ v16bf frag_combine(v8bf lo, v8bf hi) {
    v16bf f;
#pragma unroll
    for (int i = 0; i < 8; ++i) { f[i] = lo[i]; f[i + 8] = hi[i]; }
    return f;
}

// Fragment from global memory, row-major [rows, ld] bf16, k-slice k0.
__device__ __forceinline__ v16bf load_frag_g(const bf16* __restrict__ base,
                                             int ld, int row, int k0, int lane) {
    const int r = row + (lane & 15);
    const int g = lane >> 4;
    const bf16* p = base + (size_t)r * ld + k0 + g * 8;
    return frag_combine(*(const v8bf*)(p), *(const v8bf*)(p + 16));
}

// Fragment from an LDS tile with ROW_B-byte row stride (32 bf16 of data/row).
__device__ __forceinline__ v16bf load_frag_lds(const char* tile, int row, int lane) {
    const int r = row + (lane & 15);
    const int g = lane >> 4;
    const char* p = tile + r * ROW_B + g * 16;
    return frag_combine(*(const v8bf*)(p), *(const v8bf*)(p + 32));
}

__device__ __forceinline__ float sigmoidf_dev(float x) {
    return 1.0f / (1.0f + __expf(-x));
}

// ---------------------------------------------------------------------------
// TDM: issue one 2D tile DMA (tileRows x 32 bf16) global -> LDS.
// D# per CDNA5 ISA 8.3/8.4: count=1, type=2, data_size=1(2B),
// pad_enable=1, pad_interval=3 (16 dwords), pad_amount=3 (4 dwords) -> 80B rows.
// ---------------------------------------------------------------------------
#if defined(HAVE_TDM)
__device__ __forceinline__ void tdm_load_tile(unsigned lds_off,
                                              const bf16* gsrc,
                                              unsigned tileRows,
                                              unsigned tensorRows,
                                              unsigned K) {
    const unsigned long long ga = (unsigned long long)(uintptr_t)gsrc;
    v4u g0;
    g0[0] = 1u;                                            // count=1, user mode
    g0[1] = lds_off;                                       // lds_addr
    g0[2] = (unsigned)(ga & 0xFFFFFFFFu);                  // global_addr[31:0]
    g0[3] = (unsigned)((ga >> 32) & 0x01FFFFFFu) | (2u << 30); // addr[56:32] | type=2

    const unsigned d0 = (1u << 16)      // data_size = 2 bytes
                      | (1u << 20)      // pad_enable
                      | (3u << 22)      // pad_interval = 16 dwords
                      | (3u << 25);     // pad_amount   = 4 dwords
    const unsigned td0 = K;             // tensor_dim0 (k extent)
    const unsigned td1 = tensorRows;    // tensor_dim1 (row extent)
    v8i g1;
    g1[0] = (int)d0;
    g1[1] = (int)((td0 & 0xFFFFu) << 16);                  // tensor_dim0[15:0]
    g1[2] = (int)((td0 >> 16) | ((td1 & 0xFFFFu) << 16));  // dim0[31:16] | dim1[15:0]
    g1[3] = (int)((td1 >> 16) | ((unsigned)BK << 16));     // dim1[31:16] | tile_dim0
    g1[4] = (int)(tileRows & 0xFFFFu);                     // tile_dim1 (tile_dim2=0)
    g1[5] = (int)K;                                        // tensor_dim0_stride[31:0]
    g1[6] = 0;                                             // stride0[47:32] | stride1[15:0]
    g1[7] = 0;                                             // stride1[47:16]
    const v4i z4 = {0, 0, 0, 0};
#if __clang_major__ >= 23
    const v8i z8 = {0, 0, 0, 0, 0, 0, 0, 0};
    __builtin_amdgcn_tensor_load_to_lds(g0, g1, z4, z4, z8, 0);
#else
    __builtin_amdgcn_tensor_load_to_lds(g0, g1, z4, z4, 0);
#endif
}
#endif

// ---------------------------------------------------------------------------
// 0) init: zero h slot 0 (bf16) and c state (f32)
// ---------------------------------------------------------------------------
__global__ __launch_bounds__(256) void init_state_kernel(bf16* __restrict__ h0,
                                                         float* __restrict__ c) {
    int i = blockIdx.x * 256 + threadIdx.x;
    if (i < BATCH * HID_D) {
        h0[i] = (bf16)0.0f;
        c[i]  = 0.0f;
    }
}

// ---------------------------------------------------------------------------
// 1) embedding gather + f32 -> bf16 cast.  One block per (t,b) row.
// ---------------------------------------------------------------------------
__global__ __launch_bounds__(256) void embed_cast_kernel(const int*   __restrict__ tokens,
                                                         const float* __restrict__ emb,
                                                         bf16*        __restrict__ xbf) {
    const int row = blockIdx.x;
    const int tok = tokens[row];
    const float* src = emb + (size_t)tok * EMB_D;
    bf16* dst = xbf + (size_t)row * EMB_D;
    for (int j = threadIdx.x; j < EMB_D; j += 256)
        dst[j] = (bf16)src[j];
}

// ---------------------------------------------------------------------------
// 2) W [K,N] f32 row-major -> Wt [N,K] bf16 row-major (LDS tile transpose)
// ---------------------------------------------------------------------------
__global__ __launch_bounds__(256) void transpose_cast_kernel(const float* __restrict__ W,
                                                             bf16*        __restrict__ Wt,
                                                             int K, int N) {
    __shared__ float tile[32][33];
    const int nb = blockIdx.x * 32;
    const int kb = blockIdx.y * 32;
    const int tx = threadIdx.x;
    const int ty = threadIdx.y;
#pragma unroll
    for (int i = 0; i < 32; i += 8)
        tile[ty + i][tx] = W[(size_t)(kb + ty + i) * N + (nb + tx)];
    __syncthreads();
#pragma unroll
    for (int i = 0; i < 32; i += 8)
        Wt[(size_t)(nb + ty + i) * K + (kb + tx)] = (bf16)tile[tx][ty + i];
}

// ---------------------------------------------------------------------------
// 3) Big GEMM: C[M,N] f32 = A[M,K]bf16 x Bt[N,K]bf16 + bias[N]
//    Block: 8 waves, 64(M) x 256(N) tile; wave: 32x64 = 8 WMMAs per k-stage.
//    TDM DMA stages A (64x32) + B (256x32) bf16 tiles into LDS, double-buffered.
// ---------------------------------------------------------------------------
__global__ __launch_bounds__(256) void gemm_bf16_kernel(const bf16*  __restrict__ A,
                                                        const bf16*  __restrict__ Bt,
                                                        float*       __restrict__ C,
                                                        const float* __restrict__ bias,
                                                        int M, int N, int K) {
    __shared__ __attribute__((aligned(128))) char smem[2 * BUF_B];
    const int tid  = threadIdx.x;
    const int wave = tid >> 5;
    const int lane = tid & 31;
    const int mBlk = blockIdx.x * BM;
    const int nBlk = blockIdx.y * BN;
    const int mW = (wave & 1) * 32;    // wave's 32-row slab inside A tile
    const int nW = (wave >> 1) * 64;   // wave's 64-col slab inside B tile

    v8f acc[2][4];
#pragma unroll
    for (int i = 0; i < 2; ++i)
#pragma unroll
        for (int j = 0; j < 4; ++j) acc[i][j] = (v8f){};

    const int kIters = K / BK;
    const unsigned ldsBase = (unsigned)(uintptr_t)(&smem[0]);

#if defined(HAVE_TDM)
    if (tid < 32) {   // wave 0 drives the DMA engine
        tdm_load_tile(ldsBase,            A  + (size_t)mBlk * K, BM, (unsigned)M, (unsigned)K);
        tdm_load_tile(ldsBase + A_TILE_B, Bt + (size_t)nBlk * K, BN, (unsigned)N, (unsigned)K);
    }
#endif

    for (int it = 0; it < kIters; ++it) {
        const int cur = it & 1;
        const char* bufA = smem + cur * BUF_B;
        const char* bufB = bufA + A_TILE_B;

#if defined(HAVE_TDM)
        if (tid < 32) __builtin_amdgcn_s_wait_tensorcnt(0);   // current buffer landed
        __syncthreads();
        if (tid < 32 && (it + 1) < kIters) {                  // prefetch next k-slice
            const unsigned nb = ldsBase + ((it + 1) & 1) * BUF_B;
            const int k0 = (it + 1) * BK;
            tdm_load_tile(nb,            A  + (size_t)mBlk * K + k0, BM, (unsigned)M, (unsigned)K);
            tdm_load_tile(nb + A_TILE_B, Bt + (size_t)nBlk * K + k0, BN, (unsigned)N, (unsigned)K);
        }
#else
        // Fallback: cooperative 16B-chunk staging (no TDM builtin available).
        __syncthreads();
        {
            const int k0 = it * BK;
            char* dA = smem + cur * BUF_B;
            char* dB = dA + A_TILE_B;
            // A: 64 rows x 4 chunks = 256 chunks, one per thread
            {
                const int r = tid >> 2, s = tid & 3;
                *(v8bf*)(dA + r * ROW_B + s * 16) =
                    *(const v8bf*)(A + (size_t)(mBlk + r) * K + k0 + s * 8);
            }
            // B: 256 rows x 4 chunks = 1024 chunks, 4 per thread
#pragma unroll
            for (int cchunk = 0; cchunk < 4; ++cchunk) {
                const int cidx = cchunk * 256 + tid;
                const int r = cidx >> 2, s = cidx & 3;
                *(v8bf*)(dB + r * ROW_B + s * 16) =
                    *(const v8bf*)(Bt + (size_t)(nBlk + r) * K + k0 + s * 8);
            }
        }
        __syncthreads();
#endif

        // ---- compute 8 WMMAs from LDS ----
        v16bf a0 = load_frag_lds(bufA, mW,      lane);
        v16bf a1 = load_frag_lds(bufA, mW + 16, lane);
#pragma unroll
        for (int t = 0; t < 4; ++t) {
            v16bf bfr = load_frag_lds(bufB, nW + t * 16, lane);
            acc[0][t] = __builtin_amdgcn_wmma_f32_16x16x32_bf16(false, a0, false, bfr, (short)0, acc[0][t], false, false);
            acc[1][t] = __builtin_amdgcn_wmma_f32_16x16x32_bf16(false, a1, false, bfr, (short)0, acc[1][t], false, false);
        }

#if defined(HAVE_TDM)
        __syncthreads();   // everyone done reading buf before it is DMA-overwritten
#endif
    }

    // ---- epilogue: C/D layout n = lane&15, VGPR r -> m = r + 8*(lane>>4) ----
    const int nCol  = lane & 15;
    const int rBase = (lane >> 4) * 8;
#pragma unroll
    for (int s = 0; s < 2; ++s) {
#pragma unroll
        for (int t = 0; t < 4; ++t) {
            const int nc = nBlk + nW + t * 16 + nCol;
            const float bv = bias ? bias[nc] : 0.0f;
#pragma unroll
            for (int r = 0; r < 8; ++r) {
                const int m = mBlk + mW + s * 16 + rBase + r;
                C[(size_t)m * N + nc] = acc[s][t][r] + bv;
            }
        }
    }
}

// ---------------------------------------------------------------------------
// 4a) Per-step gate GEMM: gates[32,4096] = Xpre_t[32,4096] + h_prev @ Wh_t
//     Small, latency-bound: direct-load WMMA, grid = 64 blocks.
// ---------------------------------------------------------------------------
__global__ __launch_bounds__(256) void lstm_gates_kernel(const bf16*  __restrict__ h_prev,
                                                         const bf16*  __restrict__ Wh_t,
                                                         const float* __restrict__ Xpre_t,
                                                         float*       __restrict__ gates) {
    const int wave = threadIdx.x >> 5;
    const int lane = threadIdx.x & 31;
    const int m0 = (wave & 1) * 16;
    const int n0 = blockIdx.x * 64 + (wave >> 1) * 16;

    v8f acc = {};
    for (int k0 = 0; k0 < HID_D; k0 += 32) {
        v16bf a = load_frag_g(h_prev, HID_D, m0, k0, lane);
        v16bf b = load_frag_g(Wh_t,   HID_D, n0, k0, lane);
        acc = __builtin_amdgcn_wmma_f32_16x16x32_bf16(false, a, false, b, (short)0, acc, false, false);
    }

    const int nc    = n0 + (lane & 15);
    const int rBase = (lane >> 4) * 8;
#pragma unroll
    for (int r = 0; r < 8; ++r) {
        const int m = m0 + rBase + r;
        gates[(size_t)m * G4H + nc] = acc[r] + Xpre_t[(size_t)m * G4H + nc];
    }
}

// ---------------------------------------------------------------------------
// 4b) LSTM cell nonlinearity: c,h update; h written as bf16 into history.
// ---------------------------------------------------------------------------
__global__ __launch_bounds__(256) void lstm_cell_kernel(const float* __restrict__ gates,
                                                        float*       __restrict__ c,
                                                        bf16*        __restrict__ h_out) {
    const int idx = blockIdx.x * 256 + threadIdx.x;
    if (idx >= BATCH * HID_D) return;
    const int b = idx >> 10;
    const int j = idx & (HID_D - 1);
    const float* g = gates + (size_t)b * G4H;
    const float i_ = sigmoidf_dev(g[j]);
    const float f_ = sigmoidf_dev(g[j + HID_D]);
    const float gg = tanhf(g[j + 2 * HID_D]);
    const float o_ = sigmoidf_dev(g[j + 3 * HID_D]);
    const float cn = f_ * c[idx] + i_ * gg;
    c[idx] = cn;
    h_out[idx] = (bf16)(o_ * tanhf(cn));
}

// ---------------------------------------------------------------------------
// Host-side orchestration
// ---------------------------------------------------------------------------
extern "C" void kernel_launch(void* const* d_in, const int* in_sizes, int n_in,
                              void* d_out, int out_size, void* d_ws, size_t ws_size,
                              hipStream_t stream) {
    const int*   tokens = (const int*)  d_in[0];
    const float* emb    = (const float*)d_in[1];
    const float* W_x    = (const float*)d_in[2];
    const float* W_h    = (const float*)d_in[3];
    const float* b      = (const float*)d_in[4];
    const float* W_h2o  = (const float*)d_in[5];
    const float* b_o    = (const float*)d_in[6];
    float* out = (float*)d_out;

    const int MROWS = T_STEPS * BATCH;   // 4096

    // ---- workspace carve-up (256B aligned) ----
    char* ws = (char*)d_ws;
    size_t off = 0;
    auto carve = [&](size_t bytes) -> char* {
        char* p = ws + off;
        off = (off + bytes + 255) & ~(size_t)255;
        return p;
    };
    bf16*  Xbf    = (bf16*) carve((size_t)MROWS * EMB_D * sizeof(bf16));
    bf16*  Wx_t   = (bf16*) carve((size_t)G4H  * EMB_D * sizeof(bf16));
    bf16*  Wh_t   = (bf16*) carve((size_t)G4H  * HID_D * sizeof(bf16));
    bf16*  Wo_t   = (bf16*) carve((size_t)NCLS * HID_D * sizeof(bf16));
    float* Xpre   = (float*)carve((size_t)MROWS * G4H  * sizeof(float));
    bf16*  hall   = (bf16*) carve((size_t)(T_STEPS + 1) * BATCH * HID_D * sizeof(bf16));
    float* cstate = (float*)carve((size_t)BATCH * HID_D * sizeof(float));
    float* gates  = (float*)carve((size_t)BATCH * G4H  * sizeof(float));
    (void)ws_size; (void)in_sizes; (void)n_in; (void)out_size;

    // 0) zero h_0 and c
    init_state_kernel<<<(BATCH * HID_D + 255) / 256, 256, 0, stream>>>(hall, cstate);

    // 1) embedding gather + cast
    embed_cast_kernel<<<MROWS, 256, 0, stream>>>(tokens, emb, Xbf);

    // 2) weight transposes to bf16 [N,K]
    {
        dim3 blk(32, 8);
        transpose_cast_kernel<<<dim3(G4H / 32,  EMB_D / 32), blk, 0, stream>>>(W_x,   Wx_t, EMB_D, G4H);
        transpose_cast_kernel<<<dim3(G4H / 32,  HID_D / 32), blk, 0, stream>>>(W_h,   Wh_t, HID_D, G4H);
        transpose_cast_kernel<<<dim3(NCLS / 32, HID_D / 32), blk, 0, stream>>>(W_h2o, Wo_t, HID_D, NCLS);
    }

    // 3) Xpre = Xbf @ Wx_t + b   (M=4096, N=4096, K=1024)
    gemm_bf16_kernel<<<dim3(MROWS / BM, G4H / BN), 256, 0, stream>>>(
        Xbf, Wx_t, Xpre, b, MROWS, G4H, EMB_D);

    // 4) sequential LSTM over T steps
    for (int t = 0; t < T_STEPS; ++t) {
        const bf16*  h_prev = hall + (size_t)t * BATCH * HID_D;
        bf16*        h_next = hall + (size_t)(t + 1) * BATCH * HID_D;
        const float* Xpre_t = Xpre + (size_t)t * BATCH * G4H;
        lstm_gates_kernel<<<G4H / 64, 256, 0, stream>>>(h_prev, Wh_t, Xpre_t, gates);
        lstm_cell_kernel<<<(BATCH * HID_D + 255) / 256, 256, 0, stream>>>(gates, cstate, h_next);
    }

    // 5) readout: out = h[1..T] @ Wo_t + b_o   (M=4096, N=32000, K=1024)
    gemm_bf16_kernel<<<dim3(MROWS / BM, NCLS / BN), 256, 0, stream>>>(
        hall + (size_t)BATCH * HID_D, Wo_t, out, b_o, MROWS, NCLS, HID_D);
}
